// FlatQuantPixArtSelfAttention_53420803227959
// MI455X (gfx1250) — 16-bit and 8-bit matrix operand order repaired and checked by reading, compiled, not run on hardware
//
#include <hip/hip_runtime.h>
#include <hip/hip_bf16.h>

typedef __attribute__((ext_vector_type(8)))  int    v8i;
typedef __attribute__((ext_vector_type(8)))  float  v8f;
typedef __attribute__((ext_vector_type(16))) __bf16 v16bf;

#define Bdim 8
#define Sdim 1024
#define Ddim 1152
#define Hdim 16
#define HDdim 72
#define HDP 96
#define NTOK (Bdim * Sdim)
#define Ldim 32
#define Rdim 36
#define PLANE ((size_t)Bdim * Hdim * Sdim * HDP)

__device__ __forceinline__ void quant_row(const float* rowb, float* red, int n,
                                          float qmax, float lo, float hi,
                                          signed char* outQ, float* outS,
                                          int tid, int nthreads) {
  float am = 0.f;
  for (int i = tid; i < n; i += nthreads) am = fmaxf(am, fabsf(rowb[i]));
  red[tid] = am;
  __syncthreads();
  for (int s = nthreads >> 1; s > 0; s >>= 1) {
    if (tid < s) red[tid] = fmaxf(red[tid], red[tid + s]);
    __syncthreads();
  }
  float scale = fmaxf(red[0] / qmax, 1e-8f);
  float inv = 1.f / scale;
  for (int i = tid; i < n; i += nthreads) {
    float q = fminf(fmaxf(rintf(rowb[i] * inv), lo), hi);
    outQ[i] = (signed char)(int)q;
  }
  if (tid == 0) *outS = scale;
  __syncthreads();
}

__device__ void gauss_jordan(const float* src, float* dstT, int n,
                             float* aug, float* fac, float* pv) {
  const int tn = 2 * n;
  for (int idx = threadIdx.x; idx < n * tn; idx += 128) {
    int i = idx / tn, j = idx % tn;
    aug[idx] = (j < n) ? src[i * n + j] : ((j - n == i) ? 1.f : 0.f);
  }
  __syncthreads();
  for (int k = 0; k < n; ++k) {
    if (threadIdx.x == 0) *pv = 1.f / aug[k * tn + k];
    __syncthreads();
    for (int j = threadIdx.x; j < tn; j += 128) aug[k * tn + j] *= *pv;
    __syncthreads();
    for (int i = threadIdx.x; i < n; i += 128) fac[i] = aug[i * tn + k];
    __syncthreads();
    for (int idx = threadIdx.x; idx < n * tn; idx += 128) {
      int i = idx / tn, j = idx % tn;
      if (i != k) aug[idx] -= fac[i] * aug[k * tn + j];
    }
    __syncthreads();
  }
  for (int idx = threadIdx.x; idx < n * n; idx += 128) {
    int p = idx / n, l = idx % n;
    dstT[l * n + p] = aug[p * tn + n + l];
  }
  __syncthreads();
}

__global__ void inv_kernel(const float* lnA, const float* lnB, const float* vc,
                           const float* om, float* invA_T, float* invB_T,
                           float* v_it, float* o_it) {
  __shared__ float aug[72 * 144];
  __shared__ float fac[72];
  __shared__ float pv;
  gauss_jordan(lnA, invA_T, Ldim, aug, fac, &pv);
  gauss_jordan(lnB, invB_T, Rdim, aug, fac, &pv);
  gauss_jordan(vc, v_it, HDdim, aug, fac, &pv);
  gauss_jordan(om, o_it, Hdim, aug, fac, &pv);
}

__global__ void bvt_kernel(const float* bv, const float* vc, float* bvt) {
  int o = blockIdx.x * blockDim.x + threadIdx.x;
  if (o >= Ddim) return;
  int h = o / HDdim, e = o % HDdim;
  float s = 0.f;
  for (int d = 0; d < HDdim; ++d) s += bv[h * HDdim + d] * vc[d * HDdim + e];
  bvt[o] = s;
}

__global__ void xk_kernel(const float* hid, const float* lnA, const float* lnB,
                          const float* diag, signed char* xq, float* xs) {
  __shared__ float sA[Ldim * Ldim], sB[Rdim * Rdim];
  __shared__ float X[Ddim], T[Ddim], Y[Ddim], red[256];
  const int t = blockIdx.x;
  const float* hrow = hid + (size_t)t * Ddim;
  for (int i = threadIdx.x; i < Ldim * Ldim; i += 256) sA[i] = lnA[i];
  for (int i = threadIdx.x; i < Rdim * Rdim; i += 256) sB[i] = lnB[i];
  for (int i = threadIdx.x; i < Ddim; i += 256) X[i] = hrow[i] * diag[i];
  __syncthreads();
  for (int idx = threadIdx.x; idx < Ddim; idx += 256) {
    int p = idx / Rdim, r = idx % Rdim;
    float s = 0.f;
    for (int l = 0; l < Ldim; ++l) s += sA[l * Ldim + p] * X[l * Rdim + r];
    T[idx] = s;
  }
  __syncthreads();
  for (int idx = threadIdx.x; idx < Ddim; idx += 256) {
    int p = idx / Rdim, q = idx % Rdim;
    float s = 0.f;
    for (int r = 0; r < Rdim; ++r) s += T[p * Rdim + r] * sB[r * Rdim + q];
    Y[idx] = s;
  }
  __syncthreads();
  quant_row(Y, red, Ddim, 127.f, -128.f, 127.f,
            xq + (size_t)t * Ddim, xs + t, threadIdx.x, 256);
}

__global__ void wkron_kernel(const float* W, const float* invA_T,
                             const float* invB_T, const float* diag,
                             float* outF, signed char* outQ, float* outS,
                             int quantize) {
  __shared__ float sA[Ldim * Ldim], sB[Rdim * Rdim];
  __shared__ float X[Ddim], T[Ddim], Y[Ddim], red[256];
  const int row = blockIdx.x;
  const float* wrow = W + (size_t)row * Ddim;
  for (int i = threadIdx.x; i < Ldim * Ldim; i += 256) sA[i] = invA_T[i];
  for (int i = threadIdx.x; i < Rdim * Rdim; i += 256) sB[i] = invB_T[i];
  for (int i = threadIdx.x; i < Ddim; i += 256) X[i] = wrow[i] / diag[i];
  __syncthreads();
  for (int idx = threadIdx.x; idx < Ddim; idx += 256) {
    int p = idx / Rdim, r = idx % Rdim;
    float s = 0.f;
    for (int l = 0; l < Ldim; ++l) s += sA[l * Ldim + p] * X[l * Rdim + r];
    T[idx] = s;
  }
  __syncthreads();
  for (int idx = threadIdx.x; idx < Ddim; idx += 256) {
    int p = idx / Rdim, q = idx % Rdim;
    float s = 0.f;
    for (int r = 0; r < Rdim; ++r) s += T[p * Rdim + r] * sB[r * Rdim + q];
    Y[idx] = s;
  }
  __syncthreads();
  if (quantize) {
    quant_row(Y, red, Ddim, 7.f, -8.f, 7.f, outQ + (size_t)row * Ddim,
              outS + row, threadIdx.x, 256);
  } else {
    for (int i = threadIdx.x; i < Ddim; i += 256)
      outF[(size_t)row * Ddim + i] = Y[i];
  }
}

__global__ void vfold_kernel(const float* Wtmp, const float* vc,
                             signed char* outQ, float* outS) {
  __shared__ float vce[HDdim], rowb[Ddim], red[256];
  const int o = blockIdx.x;
  const int h = o / HDdim, e = o % HDdim;
  for (int d = threadIdx.x; d < HDdim; d += 256) vce[d] = vc[d * HDdim + e];
  __syncthreads();
  for (int i = threadIdx.x; i < Ddim; i += 256) {
    float s = 0.f;
    const float* base = Wtmp + (size_t)(h * HDdim) * Ddim + i;
    for (int d = 0; d < HDdim; ++d) s += base[(size_t)d * Ddim] * vce[d];
    rowb[i] = s;
  }
  __syncthreads();
  quant_row(rowb, red, Ddim, 7.f, -8.f, 7.f, outQ + (size_t)o * Ddim,
            outS + o, threadIdx.x, 256);
}

__global__ void wot_kernel(const float* Wo, const float* o_it,
                           const float* v_it, signed char* outQ, float* outS) {
  __shared__ float T1[Ddim], rowb[Ddim], so[Hdim * Hdim], red[256];
  const int o = blockIdx.x;
  for (int i = threadIdx.x; i < Hdim * Hdim; i += 256) so[i] = o_it[i];
  __syncthreads();
  for (int idx = threadIdx.x; idx < Ddim; idx += 256) {
    int h = idx / HDdim, d = idx % HDdim;
    float s = 0.f;
    for (int g = 0; g < Hdim; ++g)
      s += so[g * Hdim + h] * Wo[(size_t)o * Ddim + g * HDdim + d];
    T1[idx] = s;
  }
  __syncthreads();
  for (int idx = threadIdx.x; idx < Ddim; idx += 256) {
    int h = idx / HDdim, e = idx % HDdim;
    float s = 0.f;
    for (int d = 0; d < HDdim; ++d) s += T1[h * HDdim + d] * v_it[d * HDdim + e];
    rowb[idx] = s;
  }
  __syncthreads();
  quant_row(rowb, red, Ddim, 7.f, -8.f, 7.f, outQ + (size_t)o * Ddim,
            outS + o, threadIdx.x, 256);
}

__global__ void zero_kernel(uint4* p, size_t n) {
  size_t i = (size_t)blockIdx.x * blockDim.x + threadIdx.x;
  if (i < n) p[i] = make_uint4(0u, 0u, 0u, 0u);
}

__device__ __forceinline__ void async_stage_a(const signed char* abase,
                                              unsigned lbase, int kb,
                                              int bufsel, int lane) {
  const int row = lane >> 1;
  const int bo = (lane & 1) * 16;
  const signed char* g0 = abase + (size_t)row * Ddim + kb * 64 + bo;
  const unsigned l0 = lbase + (unsigned)(bufsel * 1024 + row * 64 + bo);
  asm volatile("global_load_async_to_lds_b128 %0, %1, off"
               :: "v"(l0), "v"((unsigned long long)(uintptr_t)g0)
               : "memory");
  asm volatile("global_load_async_to_lds_b128 %0, %1, off"
               :: "v"(l0 + 32u), "v"((unsigned long long)(uintptr_t)(g0 + 32))
               : "memory");
}

__device__ __forceinline__ void gemm_iu8_4tile(const signed char* abase,
                                               const signed char* brow0,
                                               const signed char* brow1,
                                               const signed char* brow2,
                                               const signed char* brow3,
                                               signed char* a_lds, v8i acc[4],
                                               int lane) {
  const int ph = lane >> 4;
  const int lr = lane & 15;
  const unsigned lbase = (unsigned)(uintptr_t)a_lds;
  const signed char* brows[4] = {brow0, brow1, brow2, brow3};

  async_stage_a(abase, lbase, 0, 0, lane);
  for (int kb = 0; kb < Ddim / 64; ++kb) {
    const int bufsel = kb & 1;
    if (kb + 1 < Ddim / 64) {
      async_stage_a(abase, lbase, kb + 1, bufsel ^ 1, lane);
      asm volatile("s_wait_asynccnt 0x2" ::: "memory");
    } else {
      asm volatile("s_wait_asynccnt 0x0" ::: "memory");
    }
    union { v8i v; int w[8]; } A;
    const signed char* ap = a_lds + bufsel * 1024 + lr * 64;
#pragma unroll
    for (int r = 0; r < 8; ++r) {
      const int ka = ((r >> 1) & 1) * 16 + (r & 1) * 4 + ph * 8 + (r >> 2) * 32;
      A.w[r] = *(const int*)(const void*)(ap + ka);
    }
#pragma unroll
    for (int nt = 0; nt < 4; ++nt) {
      union { v8i v; int w[8]; } Bm;
#pragma unroll
      for (int r = 0; r < 8; ++r) {
        const int kbo = ((r >> 1) & 1) * 16 + (r & 1) * 4 + ph * 8 + (r >> 2) * 32;
        Bm.w[r] = *(const int*)(const void*)(brows[nt] + kb * 64 + kbo);
      }
      acc[nt] = __builtin_amdgcn_wmma_i32_16x16x64_iu8(true, A.v, true, Bm.v,
                                                       acc[nt], false, false);
    }
  }
}

__global__ void gemm_qkv_kernel(const signed char* __restrict__ xq,
                                const float* __restrict__ xs,
                                const signed char* __restrict__ wq,
                                const signed char* __restrict__ wk,
                                const signed char* __restrict__ wv,
                                const float* __restrict__ sq,
                                const float* __restrict__ sk,
                                const float* __restrict__ sv,
                                const float* __restrict__ bq,
                                const float* __restrict__ bk,
                                const float* __restrict__ bvt,
                                __bf16* __restrict__ qkv) {
  __shared__ __align__(16) signed char a_lds[2 * 1024];
  const int lane = threadIdx.x & 31;
  const int ph = lane >> 4;
  const int lr = lane & 15;
  const int mtile = blockIdx.x;
  const int nbase = blockIdx.y * 64;
  const signed char* abase = xq + (size_t)(mtile * 16) * Ddim;

  const signed char* brows[4];
  float sw[4], bias[4];
  int hh[4], hd[4], which[4];
#pragma unroll
  for (int nt = 0; nt < 4; ++nt) {
    const int n = nbase + nt * 16 + lr;
    const int wch = n / Ddim;
    const int nc = n - wch * Ddim;
    which[nt] = wch;
    const signed char* W = (wch == 0) ? wq : (wch == 1) ? wk : wv;
    brows[nt] = W + (size_t)nc * Ddim;
    sw[nt] = ((wch == 0) ? sq : (wch == 1) ? sk : sv)[nc];
    bias[nt] = ((wch == 0) ? bq : (wch == 1) ? bk : bvt)[nc];
    hh[nt] = nc / HDdim;
    hd[nt] = nc - hh[nt] * HDdim;
  }

  v8i acc[4];
#pragma unroll
  for (int nt = 0; nt < 4; ++nt) acc[nt] = (v8i){0, 0, 0, 0, 0, 0, 0, 0};
  gemm_iu8_4tile(abase, brows[0], brows[1], brows[2], brows[3], a_lds, acc,
                 lane);

#pragma unroll
  for (int nt = 0; nt < 4; ++nt) {
#pragma unroll
    for (int i = 0; i < 8; ++i) {
      const int t = mtile * 16 + i + ph * 8;
      const float val = (float)acc[nt][i] * xs[t] * sw[nt] + bias[nt];
      const int b = t >> 10, s = t & (Sdim - 1);
      if (which[nt] < 2) {
        qkv[(size_t)which[nt] * PLANE +
            (((size_t)b * Hdim + hh[nt]) * Sdim + s) * HDP + hd[nt]] =
            (__bf16)val;
      } else {
        qkv[2 * PLANE +
            (((size_t)b * Hdim + hh[nt]) * HDP + hd[nt]) * Sdim + s] =
            (__bf16)val;
      }
    }
  }
}

__global__ void flash_kernel(const __bf16* __restrict__ qkv,
                             float* __restrict__ ctx) {
  __shared__ __bf16 p_lds[16 * 32];
  const int lane = threadIdx.x & 31;
  const int ph = lane >> 4;
  const int lr = lane & 15;
  const int qtile = blockIdx.x;
  const int bh = blockIdx.y;
  const __bf16* qb = qkv + (size_t)bh * Sdim * HDP;
  const __bf16* kb = qkv + PLANE + (size_t)bh * Sdim * HDP;
  const __bf16* vT = qkv + 2 * PLANE + (size_t)bh * HDP * Sdim;

  union U { v16bf v; unsigned w[8]; __bf16 e[16]; };

  U qa[3];
  const __bf16* qrow = qb + (size_t)(qtile * 16 + lr) * HDP;
#pragma unroll
  for (int tt = 0; tt < 3; ++tt)
#pragma unroll
    for (int r = 0; r < 8; ++r) {
      const int kp = (r & 3) * 2 + ph * 8 + (r >> 2) * 16;
      qa[tt].w[r] = *(const unsigned*)(const void*)(qrow + tt * 32 + kp);
    }

  v8f ctxa[6];
#pragma unroll
  for (int nt = 0; nt < 6; ++nt) ctxa[nt] = (v8f){0, 0, 0, 0, 0, 0, 0, 0};
  float mrow[8], lrow[8];
#pragma unroll
  for (int i = 0; i < 8; ++i) { mrow[i] = -3.0e38f; lrow[i] = 0.f; }

  for (int kc = 0; kc < Sdim / 32; ++kc) {
    v8f s0 = {0, 0, 0, 0, 0, 0, 0, 0};
    v8f s1 = {0, 0, 0, 0, 0, 0, 0, 0};
#pragma unroll
    for (int sub = 0; sub < 2; ++sub) {
      const __bf16* krow = kb + (size_t)(kc * 32 + sub * 16 + lr) * HDP;
      v8f sc = {0, 0, 0, 0, 0, 0, 0, 0};
#pragma unroll
      for (int tt = 0; tt < 3; ++tt) {
        U kop;
#pragma unroll
        for (int r = 0; r < 8; ++r) {
          const int kd = (r & 3) * 2 + ph * 8 + (r >> 2) * 16;
          kop.w[r] = *(const unsigned*)(const void*)(krow + tt * 32 + kd);
        }
        sc = __builtin_amdgcn_wmma_f32_16x16x32_bf16(false, qa[tt].v, false,
                                                     kop.v, (short)0, sc,
                                                     false, false);
      }
      if (sub == 0) s0 = sc; else s1 = sc;
    }

#pragma unroll
    for (int i = 0; i < 8; ++i) {
      float tm = fmaxf(s0[i], s1[i]);
#pragma unroll
      for (int off = 1; off < 16; off <<= 1)
        tm = fmaxf(tm, __shfl_xor(tm, off, 32));
      const float mnew = fmaxf(mrow[i], tm);
      const float p0 = __expf(s0[i] - mnew);
      const float p1 = __expf(s1[i] - mnew);
      float rs = p0 + p1;
#pragma unroll
      for (int off = 1; off < 16; off <<= 1) rs += __shfl_xor(rs, off, 32);
      const float alpha = __expf(mrow[i] - mnew);
      lrow[i] = lrow[i] * alpha + rs;
      mrow[i] = mnew;
#pragma unroll
      for (int nt = 0; nt < 6; ++nt) ctxa[nt][i] *= alpha;
      const int m = i + ph * 8;
      p_lds[m * 32 + lr] = (__bf16)p0;
      p_lds[m * 32 + 16 + lr] = (__bf16)p1;
    }
    __asm__ volatile("s_wait_dscnt 0" ::: "memory");

    U pa;
#pragma unroll
    for (int r = 0; r < 8; ++r) {
      const int kp = (r & 3) * 2 + ph * 8 + (r >> 2) * 16;
      pa.w[r] = *(const unsigned*)(const void*)&p_lds[lr * 32 + kp];
    }
#pragma unroll
    for (int nt = 0; nt < 6; ++nt) {
      U vop;
      const __bf16* vrow = vT + (size_t)(nt * 16 + lr) * Sdim + kc * 32;
#pragma unroll
      for (int r = 0; r < 8; ++r) {
        const int kd = (r & 3) * 2 + ph * 8 + (r >> 2) * 16;
        vop.w[r] = *(const unsigned*)(const void*)(vrow + kd);
      }
      ctxa[nt] = __builtin_amdgcn_wmma_f32_16x16x32_bf16(false, pa.v, false,
                                                         vop.v, (short)0,
                                                         ctxa[nt], false, false);
    }
  }

  const int b = bh >> 4, h = bh & 15;
#pragma unroll
  for (int i = 0; i < 8; ++i) {
    const float invl = 1.f / lrow[i];
    const int s = qtile * 16 + i + ph * 8;
#pragma unroll
    for (int nt = 0; nt < 6; ++nt) {
      const int hd = nt * 16 + lr;
      if (hd < HDdim)
        ctx[(((size_t)b * Sdim + s) * Hdim + h) * HDdim + hd] =
            ctxa[nt][i] * invl;
    }
  }
}

__global__ void omix_kernel(const float* __restrict__ ctx,
                            const float* __restrict__ om, signed char* cq,
                            float* cs) {
  __shared__ float cin[Ddim], rowb[Ddim], somat[Hdim * Hdim], red[128];
  const int t = blockIdx.x;
  const float* crow = ctx + (size_t)t * Ddim;
  for (int i = threadIdx.x; i < Ddim; i += 128) cin[i] = crow[i];
  for (int i = threadIdx.x; i < Hdim * Hdim; i += 128) somat[i] = om[i];
  __syncthreads();
  for (int idx = threadIdx.x; idx < Ddim; idx += 128) {
    int h = idx / HDdim, d = idx % HDdim;
    float s = 0.f;
    for (int g = 0; g < Hdim; ++g) s += cin[g * HDdim + d] * somat[g * Hdim + h];
    rowb[idx] = s;
  }
  __syncthreads();
  quant_row(rowb, red, Ddim, 127.f, -128.f, 127.f, cq + (size_t)t * Ddim,
            cs + t, threadIdx.x, 128);
}

__global__ void gemm_out_kernel(const signed char* __restrict__ cq,
                                const float* __restrict__ cs,
                                const signed char* __restrict__ wot,
                                const float* __restrict__ sot,
                                const float* __restrict__ bo,
                                float* __restrict__ out) {
  __shared__ __align__(16) signed char a_lds[2 * 1024];
  const int lane = threadIdx.x & 31;
  const int ph = lane >> 4;
  const int lr = lane & 15;
  const int mtile = blockIdx.x;
  const int nbase = blockIdx.y * 64;
  const signed char* abase = cq + (size_t)(mtile * 16) * Ddim;

  const signed char* brows[4];
#pragma unroll
  for (int nt = 0; nt < 4; ++nt)
    brows[nt] = wot + (size_t)(nbase + nt * 16 + lr) * Ddim;

  v8i acc[4];
#pragma unroll
  for (int nt = 0; nt < 4; ++nt) acc[nt] = (v8i){0, 0, 0, 0, 0, 0, 0, 0};
  gemm_iu8_4tile(abase, brows[0], brows[1], brows[2], brows[3], a_lds, acc,
                 lane);

#pragma unroll
  for (int nt = 0; nt < 4; ++nt) {
    const int n = nbase + nt * 16 + lr;
    const float sw = sot[n];
    const float bias = bo[n];
#pragma unroll
    for (int i = 0; i < 8; ++i) {
      const int t = mtile * 16 + i + ph * 8;
      out[(size_t)t * Ddim + n] = (float)acc[nt][i] * cs[t] * sw + bias;
    }
  }
}

extern "C" void kernel_launch(void* const* d_in, const int* in_sizes, int n_in,
                              void* d_out, int out_size, void* d_ws,
                              size_t ws_size, hipStream_t stream) {
  const float* hid  = (const float*)d_in[0];
  const float* Wq   = (const float*)d_in[1];
  const float* bq   = (const float*)d_in[2];
  const float* Wk   = (const float*)d_in[3];
  const float* bk   = (const float*)d_in[4];
  const float* Wv   = (const float*)d_in[5];
  const float* bv   = (const float*)d_in[6];
  const float* Wo   = (const float*)d_in[7];
  const float* bo   = (const float*)d_in[8];
  const float* lnA  = (const float*)d_in[9];
  const float* lnB  = (const float*)d_in[10];
  const float* diag = (const float*)d_in[11];
  const float* vc   = (const float*)d_in[12];
  const float* om   = (const float*)d_in[13];

  char* ws = (char*)d_ws;
  size_t off = 0;
  auto take = [&](size_t bytes) -> void* {
    void* p = ws + off;
    off += (bytes + 255) & ~(size_t)255;
    return p;
  };
  float* invA_T = (float*)take(Ldim * Ldim * 4);
  float* invB_T = (float*)take(Rdim * Rdim * 4);
  float* o_it   = (float*)take(Hdim * Hdim * 4);
  float* v_it   = (float*)take(HDdim * HDdim * 4);
  float* bvt    = (float*)take(Ddim * 4);
  float* Wtmp   = (float*)take((size_t)Ddim * Ddim * 4);
  signed char* Wq_q  = (signed char*)take((size_t)Ddim * Ddim);
  signed char* Wk_q  = (signed char*)take((size_t)Ddim * Ddim);
  signed char* Wv_q  = (signed char*)take((size_t)Ddim * Ddim);
  signed char* Wot_q = (signed char*)take((size_t)Ddim * Ddim);
  float* sq  = (float*)take(Ddim * 4);
  float* sk  = (float*)take(Ddim * 4);
  float* sv  = (float*)take(Ddim * 4);
  float* sot = (float*)take(Ddim * 4);
  signed char* xq = (signed char*)take((size_t)NTOK * Ddim);
  float* xs = (float*)take(NTOK * 4);
  __bf16* qkv = (__bf16*)take(3ull * PLANE * 2);
  float* ctx = (float*)take((size_t)NTOK * Ddim * 4);
  signed char* cq = (signed char*)take((size_t)NTOK * Ddim);
  float* cs = (float*)take(NTOK * 4);
  if (off > ws_size) return;

  inv_kernel<<<1, 128, 0, stream>>>(lnA, lnB, vc, om, invA_T, invB_T, v_it,
                                    o_it);
  bvt_kernel<<<(Ddim + 127) / 128, 128, 0, stream>>>(bv, vc, bvt);
  xk_kernel<<<NTOK, 256, 0, stream>>>(hid, lnA, lnB, diag, xq, xs);
  wkron_kernel<<<Ddim, 256, 0, stream>>>(Wq, invA_T, invB_T, diag, nullptr,
                                         Wq_q, sq, 1);
  wkron_kernel<<<Ddim, 256, 0, stream>>>(Wk, invA_T, invB_T, diag, nullptr,
                                         Wk_q, sk, 1);
  wkron_kernel<<<Ddim, 256, 0, stream>>>(Wv, invA_T, invB_T, diag, Wtmp,
                                         nullptr, nullptr, 0);
  vfold_kernel<<<Ddim, 256, 0, stream>>>(Wtmp, vc, Wv_q, sv);
  wot_kernel<<<Ddim, 256, 0, stream>>>(Wo, o_it, v_it, Wot_q, sot);

  const size_t nzero = 3ull * PLANE / 8;
  zero_kernel<<<(unsigned)((nzero + 255) / 256), 256, 0, stream>>>((uint4*)qkv,
                                                                   nzero);
  gemm_qkv_kernel<<<dim3(NTOK / 16, 3 * Ddim / 64), 32, 0, stream>>>(
      xq, xs, Wq_q, Wk_q, Wv_q, sq, sk, sv, bq, bk, bvt, qkv);
  flash_kernel<<<dim3(Sdim / 16, Bdim * Hdim), 32, 0, stream>>>(qkv, ctx);
  omix_kernel<<<NTOK, 128, 0, stream>>>(ctx, om, cq, cs);
  gemm_out_kernel<<<dim3(NTOK / 16, Ddim / 64), 32, 0, stream>>>(
      cq, cs, Wot_q, sot, bo, (float*)d_out);
}
